// GeometricAdaptiveHDA_79620103733824
// MI455X (gfx1250) — matrix operand order repaired and checked by reading, not compiled, not run on hardware
//
#include <hip/hip_runtime.h>

typedef __attribute__((ext_vector_type(16))) _Float16 v16h;
typedef __attribute__((ext_vector_type(8)))  float    v8f;

#define Hh 128
#define Ww 128
#define Bb 4
#define Cc 64
#define Gg 8
#define CgN 8
#define K2c 9
#define OO 128
#define OFFC 144          // 2*K2*G
#define MSKC 72           // K2*G
#define KTOT 576          // Cc*K2
#define NCHUNK1 18        // 576/32
#define MT1 14            // ceil(216/16) -> 224 padded rows
#define S1 592            // LDS stride (halves), 32B-aligned chunks, reduced bank conflicts
#define KG 72             // Cg*K2
#define NCHUNK2 3         // 96/32
#define S2 112            // LDS stride (halves) for deform cols

// ---------------------------------------------------------------------------
// Pack weights into WMMA A-fragment layout (f16), lane-major 32B blocks.
// A-fragment K order per half-index h (v16h element) and lane-half hi:
//   contiguous: K = chunk*32 + hi*16 + h
// ---------------------------------------------------------------------------
__global__ __launch_bounds__(256) void pack_weights(
    const float* __restrict__ w_off, const float* __restrict__ w_mask,
    const float* __restrict__ weight, _Float16* __restrict__ wcat,
    _Float16* __restrict__ wgp) {
  int blk = blockIdx.x;
  int tid = threadIdx.x;
  if (blk < MT1 * NCHUNK1) {
    int mt = blk / NCHUNK1, ch = blk % NCHUNK1;
    for (int i = tid; i < 512; i += 256) {
      int lane = i >> 4, h = i & 15;
      int m = lane & 15, hi = lane >> 4;
      int p = h >> 1, o = h & 1;
      int kb = (p < 4) ? (2 * p + o) : (16 + 2 * (p - 4) + o);
      int K = ch * 32 + hi * 16 + h;
      int row = mt * 16 + m;
      float v = 0.f;
      if (row < OFFC)             v = w_off[row * KTOT + K];       // OIHW == [row][K]
      else if (row < OFFC + MSKC) v = w_mask[(row - OFFC) * KTOT + K];
      wcat[(size_t)(mt * NCHUNK1 + ch) * 512 + i] = (_Float16)v;
    }
  } else {
    int b2 = blk - MT1 * NCHUNK1;          // 0..23
    int g = b2 / NCHUNK2, ch = b2 % NCHUNK2;
    for (int i = tid; i < 512; i += 256) {
      int lane = i >> 4, h = i & 15;
      int m = lane & 15, hi = lane >> 4;
      int p = h >> 1, o = h & 1;
      int kb = (p < 4) ? (2 * p + o) : (16 + 2 * (p - 4) + o);
      int K = ch * 32 + hi * 16 + h;
      float v = 0.f;
      if (K < KG) v = weight[(g * 16 + m) * KG + K];  // (O, Cg, 3, 3) == [row][K]
      wgp[(size_t)(g * NCHUNK2 + ch) * 512 + i] = (_Float16)v;
    }
  }
}

// ---------------------------------------------------------------------------
// Fused offset+mask 3x3 conv as implicit GEMM: C[224 x 16px] = A[224 x 576] B[576 x 16px]
// One block = 16-pixel tile; 256 threads build im2col in LDS.
// Each wave computes TWO M-tiles (wv, wv+8) with shared B-fragments and a
// double-buffered A stream: 2 independent WMMA chains hide the L2 latency.
// ---------------------------------------------------------------------------
__global__ __launch_bounds__(256) void conv_offmask(
    const float* __restrict__ x, const _Float16* __restrict__ wcat,
    const float* __restrict__ b_off, const float* __restrict__ b_mask,
    float* __restrict__ offb, float* __restrict__ maskb) {
  __shared__ __align__(32) _Float16 cols[16 * S1];
  int base = blockIdx.x * 16;
  int b = base >> 14;            // H*W = 16384
  int rem = base & 16383;
  int y = rem >> 7;
  int x0 = rem & 127;
  int tid = threadIdx.x;
  int n = tid & 15, j = tid >> 4;

  int lane = tid & 31;
  int wv = tid >> 5;
  int hi = lane >> 4;
  int nn = lane & 15;

  // Dual M-tiles per wave: mt0 in 0..7, mt1 in 8..15 (mt1 valid when < 14)
  int mt0 = wv;
  int mt1 = wv + 8;
  bool has1 = (mt1 < MT1);
  const _Float16* ap0 = wcat + (size_t)mt0 * NCHUNK1 * 512 + lane * 16;
  const _Float16* ap1 = wcat + (size_t)(has1 ? mt1 : mt0) * NCHUNK1 * 512 + lane * 16;

  // Issue first A-chunk loads before the im2col phase (latency hidden by build)
  v16h a0 = *(const v16h*)(ap0);
  v16h a1 = *(const v16h*)(ap1);

  // im2col build: (c,t) pairs strided by 16 so each 16-thread group reads 64B coalesced
  for (int it = 0; it < 36; ++it) {
    int ct = j + (it << 4);                    // 0..575 == c*9+t
    int c = ct / 9, t = ct - c * 9;
    int ky = t / 3, kx = t - ky * 3;
    int yy = y + ky - 1;
    int xx = x0 + n + kx - 1;
    float v = 0.f;
    if (yy >= 0 && yy < Hh && xx >= 0 && xx < Ww)
      v = x[((b * Cc + c) * Hh + yy) * Ww + xx];
    cols[n * S1 + ct] = (_Float16)v;
  }
  __syncthreads();

  const _Float16* colbase = &cols[nn * S1 + hi * 16];   // B frag: K contiguous per lane

  v8f acc0 = {};
  v8f acc1 = {};
#pragma unroll
  for (int ch = 0; ch < NCHUNK1 - 1; ++ch) {
    // prefetch next chunk's A fragments (independent of this chunk's WMMAs)
    v16h a0n = *(const v16h*)(ap0 + (ch + 1) * 512);
    v16h a1n = *(const v16h*)(ap1 + (ch + 1) * 512);
    v16h bf = *(const v16h*)(colbase + ch * 32);
    acc0 = __builtin_amdgcn_wmma_f32_16x16x32_f16(false, a0, false, bf,
                                                  (short)0, acc0, false, false);
    acc1 = __builtin_amdgcn_wmma_f32_16x16x32_f16(false, a1, false, bf,
                                                  (short)0, acc1, false, false);
    a0 = a0n;
    a1 = a1n;
  }
  {
    v16h bf = *(const v16h*)(colbase + (NCHUNK1 - 1) * 32);
    acc0 = __builtin_amdgcn_wmma_f32_16x16x32_f16(false, a0, false, bf,
                                                  (short)0, acc0, false, false);
    acc1 = __builtin_amdgcn_wmma_f32_16x16x32_f16(false, a1, false, bf,
                                                  (short)0, acc1, false, false);
  }

  int xcol = x0 + nn;
#pragma unroll
  for (int r = 0; r < 8; ++r) {
    // C layout: vgpr r -> M = r + 8*hi, N = lane&15
    int chan0 = mt0 * 16 + r + hi * 8;
    float v0 = acc0[r];
    if (chan0 < OFFC) {
      offb[(((size_t)b * OFFC + chan0) * Hh + y) * Ww + xcol] = v0 + b_off[chan0];
    } else if (chan0 < OFFC + MSKC) {
      int mc = chan0 - OFFC;
      float s = v0 + b_mask[mc];
      maskb[(((size_t)b * MSKC + mc) * Hh + y) * Ww + xcol] = 1.f / (1.f + __expf(-s));
    }
    if (has1) {
      int chan1 = mt1 * 16 + r + hi * 8;
      float v1 = acc1[r];
      if (chan1 < OFFC) {
        offb[(((size_t)b * OFFC + chan1) * Hh + y) * Ww + xcol] = v1 + b_off[chan1];
      } else if (chan1 < OFFC + MSKC) {
        int mc = chan1 - OFFC;
        float s = v1 + b_mask[mc];
        maskb[(((size_t)b * MSKC + mc) * Hh + y) * Ww + xcol] = 1.f / (1.f + __expf(-s));
      }
    }
  }
}

// ---------------------------------------------------------------------------
// Deformable sampling + grouped GEMM. Block = 16-pixel tile, wave = group.
// A fragments preloaded before the sampling phase (latency hidden under the
// bilinear gathers); post-barrier GEMM is register-A vs LDS-B.
// ---------------------------------------------------------------------------
__global__ __launch_bounds__(256) void deform_gemm(
    const float* __restrict__ x, const float* __restrict__ offb,
    const float* __restrict__ maskb, const _Float16* __restrict__ wgp,
    const float* __restrict__ bias, float* __restrict__ out) {
  __shared__ __align__(32) _Float16 cols[Gg][16 * S2];
  int base = blockIdx.x * 16;
  int b = base >> 14;
  int rem = base & 16383;
  int y = rem >> 7;
  int x0 = rem & 127;
  int tid = threadIdx.x;
  int lane = tid & 31;
  int g = tid >> 5;                // one wave per group
  int px = lane & 15;
  int h = lane >> 4;
  int xg = x0 + px;
  _Float16* mycols = &cols[g][0];

  // Preload all 3 A fragments now: independent of LDS build, latency hidden.
  const _Float16* ap = wgp + (size_t)g * NCHUNK2 * 512 + lane * 16;
  v16h wa0 = *(const v16h*)(ap);
  v16h wa1 = *(const v16h*)(ap + 512);
  v16h wa2 = *(const v16h*)(ap + 1024);

  // zero K padding 72..95 (chunk 2 tail)
#pragma unroll
  for (int z = 0; z < 12; ++z)
    mycols[px * S2 + 72 + h * 12 + z] = (_Float16)0.f;

  // taps split 5/4 between lane halves; reconverges before WMMA
  for (int t = h; t < K2c; t += 2) {
    int ky = t / 3, kx = t - ky * 3;
    int chO = (g * K2c + t) * 2;
    size_t obase = (((size_t)b * OFFC + chO) * Hh + y) * Ww + xg;
    float offy = offb[obase];
    float offx = offb[obase + (size_t)Hh * Ww];
    float mval = maskb[(((size_t)b * MSKC + g * K2c + t) * Hh + y) * Ww + xg];
    float py  = (float)(y + ky - 1) + offy;
    float pxf = (float)(xg + kx - 1) + offx;
    float fy0 = floorf(py), fx0 = floorf(pxf);
    float wy = py - fy0, wx = pxf - fx0;
    int y0i = (int)fy0, x0i = (int)fx0;
    int y1i = y0i + 1, x1i = x0i + 1;
    float vy0 = (y0i >= 0 && y0i < Hh) ? 1.f : 0.f;
    float vy1 = (y1i >= 0 && y1i < Hh) ? 1.f : 0.f;
    float vx0 = (x0i >= 0 && x0i < Ww) ? 1.f : 0.f;
    float vx1 = (x1i >= 0 && x1i < Ww) ? 1.f : 0.f;
    int y0c = min(max(y0i, 0), Hh - 1), y1c = min(max(y1i, 0), Hh - 1);
    int x0c = min(max(x0i, 0), Ww - 1), x1c = min(max(x1i, 0), Ww - 1);
    float w00 = (1.f - wy) * (1.f - wx) * vy0 * vx0;
    float w01 = (1.f - wy) * wx * vy0 * vx1;
    float w10 = wy * (1.f - wx) * vy1 * vx0;
    float w11 = wy * wx * vy1 * vx1;
    int i00 = y0c * Ww + x0c, i01 = y0c * Ww + x1c;
    int i10 = y1c * Ww + x0c, i11 = y1c * Ww + x1c;
#pragma unroll
    for (int c = 0; c < CgN; ++c) {
      const float* img = x + ((size_t)(b * Cc + g * CgN + c) * Hh) * Ww;
      float v = w00 * img[i00] + w01 * img[i01] + w10 * img[i10] + w11 * img[i11];
      mycols[px * S2 + c * K2c + t] = (_Float16)(v * mval);
    }
  }
  __syncthreads();

  const _Float16* colbase = &mycols[(lane & 15) * S2 + h * 16];
  v16h bf0 = *(const v16h*)(colbase);
  v16h bf1 = *(const v16h*)(colbase + 32);
  v16h bf2 = *(const v16h*)(colbase + 64);
  v8f acc = {};
  acc = __builtin_amdgcn_wmma_f32_16x16x32_f16(false, wa0, false, bf0,
                                               (short)0, acc, false, false);
  acc = __builtin_amdgcn_wmma_f32_16x16x32_f16(false, wa1, false, bf1,
                                               (short)0, acc, false, false);
  acc = __builtin_amdgcn_wmma_f32_16x16x32_f16(false, wa2, false, bf2,
                                               (short)0, acc, false, false);
#pragma unroll
  for (int r = 0; r < 8; ++r) {
    int chan = g * 16 + r + h * 8;
    out[(((size_t)b * OO + chan) * Hh + y) * Ww + x0 + (lane & 15)] = acc[r] + bias[chan];
  }
}

// ---------------------------------------------------------------------------
extern "C" void kernel_launch(void* const* d_in, const int* in_sizes, int n_in,
                              void* d_out, int out_size, void* d_ws, size_t ws_size,
                              hipStream_t stream) {
  const float* x      = (const float*)d_in[0];
  const float* w_off  = (const float*)d_in[1];
  const float* b_off  = (const float*)d_in[2];
  const float* w_mask = (const float*)d_in[3];
  const float* b_mask = (const float*)d_in[4];
  const float* weight = (const float*)d_in[5];
  const float* bias   = (const float*)d_in[6];
  float* out = (float*)d_out;

  char* ws = (char*)d_ws;
  float*    offb  = (float*)ws;                         // 4*144*128*128 f32 = 37,748,736 B
  float*    maskb = (float*)(ws + 37748736);            // 4*72*128*128 f32  = 18,874,368 B
  _Float16* wcat  = (_Float16*)(ws + 56623104);         // 14*18*512 f16     =    258,048 B
  _Float16* wgp   = (_Float16*)(ws + 56881152);         // 8*3*512 f16       =     24,576 B

  pack_weights<<<MT1 * NCHUNK1 + Gg * NCHUNK2, 256, 0, stream>>>(w_off, w_mask, weight, wcat, wgp);
  conv_offmask<<<4096, 256, 0, stream>>>(x, wcat, b_off, b_mask, offb, maskb);
  deform_gemm<<<4096, 256, 0, stream>>>(x, offb, maskb, wgp, bias, out);
}
